// WaveNet_Mel2Raw_10136122819170
// MI455X (gfx1250) — matrix operand order repaired and checked by reading, not compiled, not run on hardware
//
#include <hip/hip_runtime.h>
#include <hip/hip_bf16.h>

typedef __attribute__((ext_vector_type(16))) _Float16 v16h;
typedef __attribute__((ext_vector_type(8)))  float    v8f;

constexpr int B_  = 2;
constexpr int NM  = 80;
constexpr int F_  = 128;
constexpr int R_  = 64;     // residual channels
constexpr int S_  = 128;    // skip channels
constexpr int NC  = 256;    // post channels
constexpr int UK  = 512;
constexpr int MD  = 10;
constexpr int NB  = 30;
constexpr int L_  = 32512;
constexpr int NCOND = 96;   // cond channels padded 80 -> 96 (3 K-chunks of 32)
constexpr int KZ  = 224;    // 64 (tap t-d) + 64 (tap t) + 96 (cond)
constexpr int TT  = 64;     // time tile per workgroup

// ---------------- WMMA fragment helpers (wave32, 16x16x32 f16) ----------------

// B operand: 32 K-rows (channels) x 16 N-cols (time), from f32 [t][c] row-major.
// Lanes 0-15 hold K=0..15 for column N=lane; lanes 16-31 hold K=16..31.
template<bool RELU>
__device__ __forceinline__ v16h bfrag_from_f32(const float* act, int rowstride,
                                               int t0, int kch) {
  int l = threadIdx.x & 31;
  int n = l & 15, hs = l >> 4;
  const float* p = act + (size_t)(t0 + n) * rowstride + kch + hs * 16;
  v16h r;
#pragma unroll
  for (int j = 0; j < 16; ++j) {
    float v = p[j];
    if (RELU) v = fmaxf(v, 0.f);
    r[j] = (_Float16)v;
  }
  return r;
}

__device__ __forceinline__ v16h bfrag_from_f16(const _Float16* act, int rowstride,
                                               int t0, int kch) {
  int l = threadIdx.x & 31;
  int n = l & 15, hs = l >> 4;
  const _Float16* p = act + (size_t)(t0 + n) * rowstride + kch + hs * 16;
  v16h r;
#pragma unroll
  for (int j = 0; j < 16; ++j) r[j] = p[j];
  return r;
}

// Guarded variant: zero-fills lanes with t < 0 (used only on boundary tiles).
__device__ __forceinline__ v16h bfrag_from_f16_guard(const _Float16* act, int rowstride,
                                                     int t0, int kch) {
  int l = threadIdx.x & 31;
  int n = l & 15, hs = l >> 4;
  int t = t0 + n;
  v16h r;
  if (t >= 0) {
    const _Float16* p = act + (size_t)t * rowstride + kch + hs * 16;
#pragma unroll
    for (int j = 0; j < 16; ++j) r[j] = p[j];
  } else {
#pragma unroll
    for (int j = 0; j < 16; ++j) r[j] = (_Float16)0.f;
  }
  return r;
}

// A operand: 16 M-rows x 32 K-cols from f16 weight matrix W[M][KW].
// Lane = M row; elems 0..15 = K(hs*16 + 0..15).
__device__ __forceinline__ v16h afrag(const _Float16* W, int KW, int m0, int kb) {
  int l = threadIdx.x & 31;
  int m = l & 15, hs = l >> 4;
  const _Float16* p = W + (size_t)(m0 + m) * KW + kb + hs * 16;
  v16h a;
#pragma unroll
  for (int j = 0; j < 16; ++j) a[j] = p[j];
  return a;
}

#define WMMA_F16(a, b, c) \
  __builtin_amdgcn_wmma_f32_16x16x32_f16(false, (a), false, (b), (short)0, (c), false, false)

// ---------------- weight prep: f32 -> GEMM-ready f16 arena ----------------

__global__ __launch_bounds__(256)
void weight_prep_kernel(const float* __restrict__ wdil, const float* __restrict__ wcond,
                        const float* __restrict__ wskip, const float* __restrict__ wres,
                        const float* __restrict__ w1, const float* __restrict__ w2,
                        _Float16* __restrict__ Wz, _Float16* __restrict__ Ws,
                        _Float16* __restrict__ Wr, _Float16* __restrict__ W1h,
                        _Float16* __restrict__ W2h) {
  int i = blockIdx.x, tid = threadIdx.x;
  if (i < NB) {
    const float* wd = wdil + (size_t)i * 128 * 64 * 2;   // (2R, R, KC)
    const float* wc = wcond + (size_t)i * 128 * NM;      // (2R, NM, 1)
    _Float16* dz = Wz + (size_t)i * 128 * KZ;
    for (int idx = tid; idx < 128 * KZ; idx += 256) {
      int c = idx / KZ, k = idx % KZ;
      float v;
      if (k < 64)       v = wd[(c * 64 + k) * 2 + 0];          // tap at t-d
      else if (k < 128) v = wd[(c * 64 + (k - 64)) * 2 + 1];   // tap at t
      else { int ci = k - 128; v = (ci < NM) ? wc[c * NM + ci] : 0.f; }
      dz[idx] = (_Float16)v;
    }
    const float* wsp = wskip + (size_t)i * S_ * R_;
    _Float16* dsk = Ws + (size_t)i * S_ * R_;
    for (int idx = tid; idx < S_ * R_; idx += 256) dsk[idx] = (_Float16)wsp[idx];
    const float* wrp = wres + (size_t)i * R_ * R_;
    _Float16* drr = Wr + (size_t)i * R_ * R_;
    for (int idx = tid; idx < R_ * R_; idx += 256) drr[idx] = (_Float16)wrp[idx];
  } else if (i == NB) {
    for (int idx = tid; idx < NC * S_; idx += 256) W1h[idx] = (_Float16)w1[idx];
  } else {
    for (int idx = tid; idx < NC * NC; idx += 256) W2h[idx] = (_Float16)w2[idx];
  }
}

// ---------------- cond upsampler (transposed conv, 2 taps per output) ----------------

__global__ __launch_bounds__(128)
void cond_kernel(const float* __restrict__ mels, const float* __restrict__ mel_w,
                 const float* __restrict__ mel_b, _Float16* __restrict__ cond16, int L) {
  __shared__ float sm0[NM], sm1[NM];
  int bt = blockIdx.x;
  int b = bt / L, t = bt % L;
  // taps: k0 = 255 - (t mod 256) hits frame m0 = t>>8 ; k1 = k0+256 hits m0+1.
  int k0 = 255 - (t & 255);
  int m0 = t >> 8;
  int tid = threadIdx.x;
  if (tid < NM) {
    sm0[tid] = mels[((size_t)b * NM + tid) * F_ + m0];
    sm1[tid] = mels[((size_t)b * NM + tid) * F_ + m0 + 1];
  }
  __syncthreads();
  if (tid < NCOND) {
    float acc = 0.f;
    if (tid < NM) {
      acc = mel_b[tid];
      const float* wr = mel_w + (size_t)tid * NM * UK;   // (NM, NM, UK)
      int k1 = k0 + 256;
      for (int ci = 0; ci < NM; ++ci)
        acc += wr[ci * UK + k0] * sm0[ci] + wr[ci * UK + k1] * sm1[ci];
    }
    cond16[((size_t)b * L + t) * NCOND + tid] = (_Float16)acc;  // pad ch 80..95 = 0
  }
}

// ---------------- init: res0 = input_w*x + input_b (f32 + f16 copy); skip = 0 -----

__global__ __launch_bounds__(256)
void init_kernel(const float* __restrict__ x, const float* __restrict__ iw,
                 const float* __restrict__ ib, float* __restrict__ res,
                 _Float16* __restrict__ res16, float* __restrict__ skip, int L) {
  size_t LB = (size_t)B_ * L;
  size_t nres = LB * R_, nskip = LB * S_;
  for (size_t idx = (size_t)blockIdx.x * 256 + threadIdx.x; idx < nres + nskip;
       idx += (size_t)gridDim.x * 256) {
    if (idx < nres) {
      size_t bt = idx / R_;
      int c = (int)(idx % R_);
      float v = iw[c] * x[bt] + ib[c];
      res[idx] = v;
      res16[idx] = (_Float16)v;
    } else {
      skip[idx - nres] = 0.f;
    }
  }
}

// ---------------- one WaveNet residual block ----------------

__global__ __launch_bounds__(256)
void wavenet_block_kernel(const float* __restrict__ res_in,
                          const _Float16* __restrict__ res16_in,
                          float* __restrict__ res_out,
                          _Float16* __restrict__ res16_out,
                          float* __restrict__ skip, const _Float16* __restrict__ cond16,
                          const _Float16* __restrict__ Wz, const float* __restrict__ bdil,
                          const float* __restrict__ bcond,
                          const _Float16* __restrict__ Ws, const float* __restrict__ bskip,
                          const _Float16* __restrict__ Wr, const float* __restrict__ bres,
                          int d, int L) {
  __shared__ float    sZB[128];
  __shared__ float    sSB[S_];
  __shared__ float    sRB[R_];
  __shared__ float    sZ[TT * 132];     // z tile, time-major, padded stride
  __shared__ _Float16 sH[TT * 72];      // gated h tile, time-major f16

  int tid = threadIdx.x;
  if (tid < 128) sZB[tid] = bdil[tid] + bcond[tid];
  if (tid < S_)  sSB[tid] = bskip[tid];
  if (tid < R_)  sRB[tid] = bres[tid];
  __syncthreads();

  int ntiles = L / TT;
  int b  = blockIdx.x / ntiles;
  int t0 = (blockIdx.x % ntiles) * TT;
  const float*    resI   = res_in    + (size_t)b * L * R_;
  const _Float16* res16I = res16_in  + (size_t)b * L * R_;
  float*          resO   = res_out   + (size_t)b * L * R_;
  _Float16*       res16O = res16_out + (size_t)b * L * R_;
  float*          skipP  = skip      + (size_t)b * L * S_;
  const _Float16* condP  = cond16    + (size_t)b * L * NCOND;

  int w = tid >> 5, l = tid & 31, hs = l >> 4, nn = l & 15;

  // phase 1: z(128 x 64) = Wz @ [res16(t-d); res16(t); cond(t)] + bias
  int m0 = w * 16;
  v16h az[7];
#pragma unroll
  for (int kc = 0; kc < 7; ++kc) az[kc] = afrag(Wz, KZ, m0, kc * 32);

#pragma unroll
  for (int nt = 0; nt < 4; ++nt) {
    int n0 = t0 + nt * 16;
    int nd = n0 - d;          // wave-uniform
    v8f acc;
#pragma unroll
    for (int r = 0; r < 8; ++r) acc[r] = sZB[m0 + r + 8 * hs];
    v16h bf;
    if (nd >= 0) {            // uniform branch: bulk of tiles take the fast path
      bf = bfrag_from_f16(res16I, R_, nd, 0);
      acc = WMMA_F16(az[0], bf, acc);
      bf = bfrag_from_f16(res16I, R_, nd, 32);
      acc = WMMA_F16(az[1], bf, acc);
    } else {
      bf = bfrag_from_f16_guard(res16I, R_, nd, 0);
      acc = WMMA_F16(az[0], bf, acc);
      bf = bfrag_from_f16_guard(res16I, R_, nd, 32);
      acc = WMMA_F16(az[1], bf, acc);
    }
    bf = bfrag_from_f16(res16I, R_, n0, 0);
    acc = WMMA_F16(az[2], bf, acc);
    bf = bfrag_from_f16(res16I, R_, n0, 32);
    acc = WMMA_F16(az[3], bf, acc);
#pragma unroll
    for (int kc = 0; kc < 3; ++kc) {
      bf = bfrag_from_f16(condP, NCOND, n0, kc * 32);
      acc = WMMA_F16(az[4 + kc], bf, acc);
    }
#pragma unroll
    for (int r = 0; r < 8; ++r)
      sZ[(nt * 16 + nn) * 132 + m0 + r + 8 * hs] = acc[r];
  }
  __syncthreads();

  // phase 2: h = tanh(f) * sigmoid(g)
  for (int e = tid; e < TT * R_; e += 256) {
    int t = e >> 6, c = e & 63;
    float f = sZ[t * 132 + c];
    float g = sZ[t * 132 + 64 + c];
    float hh = tanhf(f) * (1.f / (1.f + __expf(-g)));
    sH[t * 72 + c] = (_Float16)hh;
  }
  __syncthreads();

  // phase 3: skip += Ws @ h + bskip ; res_out = res_in + Wr @ h + bres
  for (int tile = w; tile < 48; tile += 8) {
    if (tile < 32) {
      int m = (tile >> 2) * 16, nt = tile & 3, n0 = t0 + nt * 16;
      v8f acc;
#pragma unroll
      for (int r = 0; r < 8; ++r) {
        int ch = m + r + 8 * hs;
        acc[r] = skipP[(size_t)(n0 + nn) * S_ + ch] + sSB[ch];
      }
#pragma unroll
      for (int kc = 0; kc < 2; ++kc) {
        v16h a  = afrag(Ws, R_, m, kc * 32);
        v16h bf = bfrag_from_f16(sH, 72, nt * 16, kc * 32);
        acc = WMMA_F16(a, bf, acc);
      }
#pragma unroll
      for (int r = 0; r < 8; ++r) {
        int ch = m + r + 8 * hs;
        skipP[(size_t)(n0 + nn) * S_ + ch] = acc[r];
      }
    } else {
      int q = tile - 32;
      int m = (q >> 2) * 16, nt = q & 3, n0 = t0 + nt * 16;
      v8f acc;
#pragma unroll
      for (int r = 0; r < 8; ++r) {
        int ch = m + r + 8 * hs;
        acc[r] = resI[(size_t)(n0 + nn) * R_ + ch] + sRB[ch];
      }
#pragma unroll
      for (int kc = 0; kc < 2; ++kc) {
        v16h a  = afrag(Wr, R_, m, kc * 32);
        v16h bf = bfrag_from_f16(sH, 72, nt * 16, kc * 32);
        acc = WMMA_F16(a, bf, acc);
      }
#pragma unroll
      for (int r = 0; r < 8; ++r) {
        int ch = m + r + 8 * hs;
        size_t off = (size_t)(n0 + nn) * R_ + ch;
        resO[off] = acc[r];
        res16O[off] = (_Float16)acc[r];
      }
    }
  }
}

// ---------------- post-net ----------------

// y1h = relu( W1 @ relu(skip) + b1 ) stored f16 time-major (B, L, 256)
__global__ __launch_bounds__(256)
void post1_kernel(const float* __restrict__ skip, const _Float16* __restrict__ W1h,
                  const float* __restrict__ b1, _Float16* __restrict__ y1h, int L) {
  __shared__ float sB[NC];
  int tid = threadIdx.x;
  if (tid < NC) sB[tid] = b1[tid];
  __syncthreads();
  int ntiles = L / TT;
  int b  = blockIdx.x / ntiles;
  int t0 = (blockIdx.x % ntiles) * TT;
  const float* sk = skip + (size_t)b * L * S_;
  _Float16* y = y1h + (size_t)b * L * NC;
  int w = tid >> 5, l = tid & 31, hs = l >> 4, nn = l & 15;
  for (int tile = w; tile < 64; tile += 8) {   // 16 m-tiles x 4 n-tiles
    int m = (tile >> 2) * 16, nt = tile & 3, n0 = t0 + nt * 16;
    v8f acc;
#pragma unroll
    for (int r = 0; r < 8; ++r) acc[r] = sB[m + r + 8 * hs];
#pragma unroll
    for (int kc = 0; kc < 4; ++kc) {
      v16h a  = afrag(W1h, S_, m, kc * 32);
      v16h bf = bfrag_from_f32<true>(sk, S_, n0, kc * 32);   // relu(skip)
      acc = WMMA_F16(a, bf, acc);
    }
#pragma unroll
    for (int r = 0; r < 8; ++r)   // store relu'd for post2's B operand
      y[(size_t)(n0 + nn) * NC + m + r + 8 * hs] = (_Float16)fmaxf(acc[r], 0.f);
  }
}

// out = W2 @ y1h + b2 ; channel-major (B, NC, L) f32 to match reference output
__global__ __launch_bounds__(256)
void post2_kernel(const _Float16* __restrict__ y1h, const _Float16* __restrict__ W2h,
                  const float* __restrict__ b2, float* __restrict__ out, int L) {
  __shared__ float sB[NC];
  int tid = threadIdx.x;
  if (tid < NC) sB[tid] = b2[tid];
  __syncthreads();
  int ntiles = L / TT;
  int b  = blockIdx.x / ntiles;
  int t0 = (blockIdx.x % ntiles) * TT;
  const _Float16* yy = y1h + (size_t)b * L * NC;
  float* op = out + (size_t)b * NC * L;
  int w = tid >> 5, l = tid & 31, hs = l >> 4, nn = l & 15;
  for (int tile = w; tile < 64; tile += 8) {
    int m = (tile >> 2) * 16, nt = tile & 3, n0 = t0 + nt * 16;
    v8f acc;
#pragma unroll
    for (int r = 0; r < 8; ++r) acc[r] = sB[m + r + 8 * hs];
#pragma unroll
    for (int kc = 0; kc < 8; ++kc) {
      v16h a  = afrag(W2h, NC, m, kc * 32);
      v16h bf = bfrag_from_f16(yy, NC, n0, kc * 32);
      acc = WMMA_F16(a, bf, acc);
    }
#pragma unroll
    for (int r = 0; r < 8; ++r) {
      int ch = m + r + 8 * hs;
      op[(size_t)ch * L + n0 + nn] = acc[r];
    }
  }
}

// ---------------- host launcher ----------------

extern "C" void kernel_launch(void* const* d_in, const int* in_sizes, int n_in,
                              void* d_out, int out_size, void* d_ws, size_t ws_size,
                              hipStream_t stream) {
  (void)in_sizes; (void)n_in; (void)out_size; (void)ws_size;
  const float* x       = (const float*)d_in[0];
  const float* mels    = (const float*)d_in[1];
  const float* input_w = (const float*)d_in[2];
  const float* input_b = (const float*)d_in[3];
  const float* mel_w   = (const float*)d_in[4];
  const float* mel_b   = (const float*)d_in[5];
  const float* bdilw   = (const float*)d_in[6];
  const float* bdilb   = (const float*)d_in[7];
  const float* bcondw  = (const float*)d_in[8];
  const float* bcondb  = (const float*)d_in[9];
  const float* bskipw  = (const float*)d_in[10];
  const float* bskipb  = (const float*)d_in[11];
  const float* bresw   = (const float*)d_in[12];
  const float* bresb   = (const float*)d_in[13];
  const float* w1      = (const float*)d_in[14];
  const float* b1      = (const float*)d_in[15];
  const float* w2      = (const float*)d_in[16];
  const float* b2      = (const float*)d_in[17];

  const int L = L_;
  size_t LB = (size_t)B_ * L;
  char* p = (char*)d_ws;
  auto alloc = [&](size_t bytes) {
    char* r = p;
    p += (bytes + 255) & ~(size_t)255;
    return r;
  };
  // region A (dead after block loop; aliased by y1h afterwards)
  float*    resA    = (float*)alloc(LB * R_ * sizeof(float));
  float*    resB    = (float*)alloc(LB * R_ * sizeof(float));
  _Float16* res16A  = (_Float16*)alloc(LB * R_ * sizeof(_Float16));
  _Float16* res16B  = (_Float16*)alloc(LB * R_ * sizeof(_Float16));
  _Float16* cond16  = (_Float16*)alloc(LB * NCOND * sizeof(_Float16));
  _Float16* Wz      = (_Float16*)alloc((size_t)NB * 128 * KZ * sizeof(_Float16));
  _Float16* Wr      = (_Float16*)alloc((size_t)NB * R_ * R_ * sizeof(_Float16));
  // persistent region
  float*    skip    = (float*)alloc(LB * S_ * sizeof(float));
  _Float16* Ws      = (_Float16*)alloc((size_t)NB * S_ * R_ * sizeof(_Float16));
  _Float16* W1h     = (_Float16*)alloc((size_t)NC * S_ * sizeof(_Float16));
  _Float16* W2h     = (_Float16*)alloc((size_t)NC * NC * sizeof(_Float16));
  // y1h (B, L, 256) f16 = 33,292,288 B aliases resA+resB — only live after blocks.
  _Float16* y1h     = (_Float16*)d_ws;

  int nt_tiles = (L / TT) * B_;   // 508 * 2 = 1016 workgroups

  weight_prep_kernel<<<NB + 2, 256, 0, stream>>>(bdilw, bcondw, bskipw, bresw, w1, w2,
                                                 Wz, Ws, Wr, W1h, W2h);
  cond_kernel<<<B_ * L, 128, 0, stream>>>(mels, mel_w, mel_b, cond16, L);
  {
    size_t total = LB * (R_ + S_);
    int blocks = (int)((total + 255) / 256);
    init_kernel<<<blocks, 256, 0, stream>>>(x, input_w, input_b, resA, res16A, skip, L);
  }

  const float*    rin  = resA;
  const _Float16* r16i = res16A;
  float*          rout = resB;
  _Float16*       r16o = res16B;
  for (int i = 0; i < NB; ++i) {
    int d = 1 << (i % MD);
    wavenet_block_kernel<<<nt_tiles, 256, 0, stream>>>(
        rin, r16i, rout, r16o, skip, cond16,
        Wz + (size_t)i * 128 * KZ,
        bdilb + (size_t)i * 128, bcondb + (size_t)i * 128,
        Ws + (size_t)i * S_ * R_, bskipb + (size_t)i * S_,
        Wr + (size_t)i * R_ * R_, bresb + (size_t)i * R_,
        d, L);
    const float* tf = rin; rin = rout; rout = (float*)tf;
    const _Float16* th = r16i; r16i = r16o; r16o = (_Float16*)th;
  }

  post1_kernel<<<nt_tiles, 256, 0, stream>>>(skip, W1h, b1, y1h, L);
  post2_kernel<<<nt_tiles, 256, 0, stream>>>(y1h, W2h, b2, (float*)d_out, L);
}
